// ENGINEComponent_87205015978354
// MI455X (gfx1250) — matrix operand order repaired and checked by reading, not compiled, not run on hardware
//
#include <hip/hip_runtime.h>
#include <hip/hip_bf16.h>

// ---------------------------------------------------------------------------
// Problem constants (from reference)
// ---------------------------------------------------------------------------
#define N_NODES 100000
#define N_EDGES 3200000
#define BATCH   1024
#define DLM     1024
#define KDIM    128
#define HDIM    256
#define CDIM    40
#define LLAYERS 2
#define LN_EPS  1e-5f
#define INV_T   10.0f          // 1 / T, T = 0.1

typedef __attribute__((ext_vector_type(16))) _Float16     v16h;
typedef __attribute__((ext_vector_type(2)))  _Float16     v2h;
typedef __attribute__((ext_vector_type(8)))  float        v8f;
typedef __attribute__((ext_vector_type(4)))  unsigned int v4u;
typedef __attribute__((ext_vector_type(8)))  unsigned int v8u;

// Pack two fp32 into one u32 of f16 halves (lowers to v_cvt_pk_f16_f32).
__device__ __forceinline__ unsigned int pack2(float x0, float x1) {
  v2h t;
  t[0] = (_Float16)x0;
  t[1] = (_Float16)x1;
  return __builtin_bit_cast(unsigned int, t);
}

// ---------------------------------------------------------------------------
// A fragment (16x32 f16) from an LDS row of packed f16 pairs.
// Layout: lane half h: VGPR j (0..7) -> K = k0+16h+2j (contiguous order).
// One u32 = (K even, K odd) pair.
// => 8 consecutive u32 at (k0+16h)/2.  Two ds_load_b128.
// ---------------------------------------------------------------------------
__device__ __forceinline__ v16h lds_a_frag(const unsigned int* row, int k0, int half) {
  const unsigned int* p = row + ((k0 + (half << 4)) >> 1);
  v4u lo = *(const v4u*)p;
  v4u hi = *(const v4u*)(p + 4);
  v8u r  = __builtin_shufflevector(lo, hi, 0, 1, 2, 3, 4, 5, 6, 7);
  return __builtin_bit_cast(v16h, r);
}

// ---------------------------------------------------------------------------
// B fragment (32x16 f16) from pre-packed weights.
// Bp index ((kt*F + c)*16 + u), u = 8*halfK + j packs (K = 32kt+16halfK+2j, +1).
// A lane's fragment = 8 consecutive u32 -> two global_load_b128.
// ---------------------------------------------------------------------------
__device__ __forceinline__ v16h load_b_packed(const unsigned int* __restrict__ Bp,
                                              int F, int k0, int col0, int lane) {
  const int half = lane >> 4;
  const int c    = col0 + (lane & 15);
  const unsigned int* p = Bp + ((size_t)(k0 >> 5) * F + c) * 16 + (half << 3);
  v4u lo = *(const v4u*)p;
  v4u hi = *(const v4u*)(p + 4);
  v8u r  = __builtin_shufflevector(lo, hi, 0, 1, 2, 3, 4, 5, 6, 7);
  return __builtin_bit_cast(v16h, r);
}

// ---------------------------------------------------------------------------
// Pre-pack a row-major fp32 weight W[Kin,F] into WMMA-B fragment order (f16).
// ---------------------------------------------------------------------------
__global__ void pack_b_kernel(const float* __restrict__ W,
                              unsigned int* __restrict__ Bp, int Kin, int F) {
  const int i = blockIdx.x * blockDim.x + threadIdx.x;
  const int total = (Kin >> 5) * F * 16;
  if (i >= total) return;
  const int u  = i & 15;
  const int c  = (i >> 4) % F;
  const int kt = (i >> 4) / F;
  const int hk = u >> 3;
  const int j  = u & 7;
  const int k  = kt * 32 + hk * 16 + 2 * j;
  Bp[i] = pack2(W[(size_t)k * F + c], W[(size_t)(k + 1) * F + c]);
}

// ---------------------------------------------------------------------------
// Fused projector: h = Linear(X,W1,b1) -> LayerNorm -> ReLU -> Linear(W2,b2)
//                  -> (layer>0) sigmoid-blend with `last`
// Block: 16 rows x 128 cols, 8 wave32s, each wave owns a 16x16 tile.
// X row-block staged once into LDS as packed f16; weights pre-packed.
// ---------------------------------------------------------------------------
__global__ __launch_bounds__(256) void proj_fused_kernel(
    const float* __restrict__ X,            // [N, 1024]
    const unsigned int* __restrict__ W1p,   // packed [1024,128]
    const float* __restrict__ b1,
    const float* __restrict__ lng,
    const float* __restrict__ lnb,
    const unsigned int* __restrict__ W2p,   // packed [128,128]
    const float* __restrict__ b2,
    const float* __restrict__ alphas,
    int layer,
    const float* __restrict__ last,         // [N,128] (read when layer > 0)
    float* __restrict__ Hout)               // [N,128]
{
  __shared__ unsigned int aTu[16][DLM / 2 + 4];   // X tile as packed f16 (33 KB)
  __shared__ float        t1[16][KDIM + 4];       // post-GEMM1 tile
  __shared__ float        rowm[16], rowi[16];     // LN stats
  __shared__ unsigned int a2u[16][KDIM / 2 + 4];  // LN+ReLU tile, packed f16

  const int tid  = threadIdx.x;
  const int wave = tid >> 5;
  const int lane = tid & 31;
  const int half = lane >> 4;
  const int cl   = lane & 15;
  const int row0 = blockIdx.x * 16;               // N % 16 == 0, no tail
  const int col0 = wave * 16;

  // ---- Stage 0: cooperative load X[row0:row0+16, :] -> LDS f16 ----
  for (int idx = tid; idx < 16 * (DLM / 4); idx += 256) {
    const int r = idx / (DLM / 4);
    const int q = idx % (DLM / 4);
    const float4 v = *(const float4*)(X + (size_t)(row0 + r) * DLM + q * 4);
    aTu[r][q * 2]     = pack2(v.x, v.y);
    aTu[r][q * 2 + 1] = pack2(v.z, v.w);
    if (idx + 256 < 16 * (DLM / 4))
      __builtin_prefetch(X + (size_t)(row0 + (idx + 256) / (DLM / 4)) * DLM +
                             ((idx + 256) % (DLM / 4)) * 4, 0, 1);
  }
  __syncthreads();

  // ---- Stage A: T1 = X @ W1 (K = 1024) ----
  v8f acc = {};
  const unsigned int* arow = &aTu[lane & 15][0];
  for (int k0 = 0; k0 < DLM; k0 += 32) {
    v16h a = lds_a_frag(arow, k0, half);
    v16h b = load_b_packed(W1p, KDIM, k0, col0, lane);
    acc = __builtin_amdgcn_wmma_f32_16x16x32_f16(false, a, false, b,
                                                 (short)0, acc, false, false);
  }
  {
    const float bias = b1[col0 + cl];
#pragma unroll
    for (int r = 0; r < 8; ++r)
      t1[r + half * 8][col0 + cl] = acc[r] + bias;
  }
  __syncthreads();

  // ---- LayerNorm stats (one thread per row) ----
  if (tid < 16) {
    float s = 0.f, s2 = 0.f;
    for (int c = 0; c < KDIM; ++c) {
      const float v = t1[tid][c];
      s += v; s2 += v * v;
    }
    const float m   = s * (1.0f / KDIM);
    const float var = s2 * (1.0f / KDIM) - m * m;
    rowm[tid] = m;
    rowi[tid] = rsqrtf(var + LN_EPS);
  }
  __syncthreads();

  // ---- Apply LN -> ReLU -> packed f16 into LDS A2 ----
  for (int idx = tid; idx < 16 * (KDIM / 2); idx += 256) {
    const int r  = idx / (KDIM / 2);
    const int c2 = (idx % (KDIM / 2)) * 2;
    float v0 = (t1[r][c2]     - rowm[r]) * rowi[r] * lng[c2]     + lnb[c2];
    float v1 = (t1[r][c2 + 1] - rowm[r]) * rowi[r] * lng[c2 + 1] + lnb[c2 + 1];
    a2u[r][c2 / 2] = pack2(fmaxf(v0, 0.f), fmaxf(v1, 0.f));
  }
  __syncthreads();

  // ---- Stage B: T2 = A2 @ W2 (K = 128) ----
  v8f acc2 = {};
  const unsigned int* arow2 = &a2u[lane & 15][0];
#pragma unroll
  for (int k0 = 0; k0 < KDIM; k0 += 32) {
    v16h a = lds_a_frag(arow2, k0, half);
    v16h b = load_b_packed(W2p, KDIM, k0, col0, lane);
    acc2 = __builtin_amdgcn_wmma_f32_16x16x32_f16(false, a, false, b,
                                                  (short)0, acc2, false, false);
  }

  // ---- Epilogue: +b2, optional sigmoid blend, write h ----
  float ablend = 1.f;
  if (layer > 0) {
    const float x = alphas[layer] * INV_T;
    ablend = 1.f / (1.f + __expf(-x));
  }
  const int col   = col0 + cl;
  const float b2v = b2[col];
#pragma unroll
  for (int r = 0; r < 8; ++r) {
    const int row = row0 + r + half * 8;
    float v = acc2[r] + b2v;
    if (layer > 0)
      v = v * ablend + last[(size_t)row * KDIM + col] * (1.f - ablend);
    Hout[(size_t)row * KDIM + col] = v;
  }
}

// ---------------------------------------------------------------------------
// WMMA GEMM: Y[N,F] = (RELU ? relu(X) : X)[N,KIN] @ W[KIN,F]  (W pre-packed)
// ---------------------------------------------------------------------------
template <int KIN, int F, bool RELU>
__global__ __launch_bounds__(256) void gemm_wmma_kernel(
    const float* __restrict__ X,
    const unsigned int* __restrict__ Wp,
    float* __restrict__ Y)
{
  __shared__ unsigned int aTu[16][KIN / 2 + 4];

  const int tid  = threadIdx.x;
  const int wave = tid >> 5;
  const int lane = tid & 31;
  const int half = lane >> 4;
  const int cl   = lane & 15;
  const int row0 = blockIdx.x * 16;

  // Cooperative stage of X row-block (fp32 -> packed f16), relu fused on input.
  for (int idx = tid; idx < 16 * (KIN / 4); idx += 256) {
    const int r = idx / (KIN / 4);
    const int q = idx % (KIN / 4);
    float4 v = *(const float4*)(X + (size_t)(row0 + r) * KIN + q * 4);
    if (RELU) {
      v.x = fmaxf(v.x, 0.f); v.y = fmaxf(v.y, 0.f);
      v.z = fmaxf(v.z, 0.f); v.w = fmaxf(v.w, 0.f);
    }
    aTu[r][q * 2]     = pack2(v.x, v.y);
    aTu[r][q * 2 + 1] = pack2(v.z, v.w);
  }
  __syncthreads();

  const unsigned int* arow = &aTu[lane & 15][0];
#pragma unroll
  for (int c0 = wave * 16; c0 < F; c0 += 128) {
    v8f acc = {};
#pragma unroll
    for (int k0 = 0; k0 < KIN; k0 += 32) {
      v16h a = lds_a_frag(arow, k0, half);
      v16h b = load_b_packed(Wp, F, k0, c0, lane);
      acc = __builtin_amdgcn_wmma_f32_16x16x32_f16(false, a, false, b,
                                                   (short)0, acc, false, false);
    }
#pragma unroll
    for (int r = 0; r < 8; ++r)
      Y[(size_t)(row0 + r + half * 8) * F + c0 + cl] = acc[r];
  }
}

// ---------------------------------------------------------------------------
// Graph / elementwise kernels
// ---------------------------------------------------------------------------
__global__ void fill_kernel(float* __restrict__ p, float v, long long n) {
  long long i = (long long)blockIdx.x * blockDim.x + threadIdx.x;
  if (i < n) p[i] = v;
}

__global__ void deg_kernel(const int* __restrict__ dst, float* __restrict__ deg,
                           long long E) {
  long long e = (long long)blockIdx.x * blockDim.x + threadIdx.x;
  if (e < E) atomicAdd(&deg[dst[e]], 1.0f);
}

__global__ void norm_kernel(const int* __restrict__ src, const int* __restrict__ dst,
                            const float* __restrict__ deg, float* __restrict__ norm,
                            long long E) {
  long long e = (long long)blockIdx.x * blockDim.x + threadIdx.x;
  if (e < E) norm[e] = rsqrtf(deg[src[e]]) * rsqrtf(deg[dst[e]]);
}

__global__ void selfw_kernel(const float* __restrict__ deg, float* __restrict__ sw,
                             int n) {
  int i = blockIdx.x * blockDim.x + threadIdx.x;
  if (i < n) sw[i] = 1.0f / deg[i];
}

// out[n,f] = hw[n,f] * self_w[n] + b[f]   (F compile-time -> shifts, no idiv)
template <int F>
__global__ void agg_init_kernel(const float* __restrict__ hw,
                                const float* __restrict__ sw,
                                const float* __restrict__ b,
                                float* __restrict__ out, long long total) {
  long long i = (long long)blockIdx.x * blockDim.x + threadIdx.x;
  if (i >= total) return;
  const long long n = i / F;          // constant F -> shift
  const int f = (int)(i % F);
  out[i] = hw[i] * sw[n] + b[f];
}

// One wave per edge: out[dst] += norm[e] * hw[src].
// src/dst/norm loaded wave-uniform (one request); features coalesced float4.
template <int F>
__global__ __launch_bounds__(256) void edge_scatter_kernel(
    const int* __restrict__ src, const int* __restrict__ dst,
    const float* __restrict__ norm, const float* __restrict__ hw,
    float* __restrict__ out, long long E) {
  const long long w = (long long)blockIdx.x * 8 + (threadIdx.x >> 5);
  if (w >= E) return;
  const int lane = threadIdx.x & 31;
  const int s    = src[w];
  const int d    = dst[w];
  const float nr = norm[w];
#pragma unroll
  for (int f0 = 0; f0 < F; f0 += 128) {
    const int f = f0 + lane * 4;
    const float4 v = *(const float4*)(hw + (size_t)s * F + f);
    float* o = out + (size_t)d * F + f;
    atomicAdd(o + 0, v.x * nr);
    atomicAdd(o + 1, v.y * nr);
    atomicAdd(o + 2, v.z * nr);
    atomicAdd(o + 3, v.w * nr);
  }
}

__global__ void cnt_kernel(const int* __restrict__ src, const int* __restrict__ dst,
                           float* __restrict__ cnt, long long E) {
  long long e = (long long)blockIdx.x * blockDim.x + threadIdx.x;
  if (e >= E) return;
  atomicAdd(&cnt[src[e]], 1.0f);
  atomicAdd(&cnt[dst[e]], 1.0f);
}

// One wave per edge: sum_n[src] += last[dst]; sum_n[dst] += last[src].
__global__ __launch_bounds__(256) void pool_scatter_kernel(
    const int* __restrict__ src, const int* __restrict__ dst,
    const float* __restrict__ last, float* __restrict__ sum_n, long long E) {
  const long long w = (long long)blockIdx.x * 8 + (threadIdx.x >> 5);
  if (w >= E) return;
  const int lane = threadIdx.x & 31;
  const int s = src[w];
  const int d = dst[w];
  const int f = lane * 4;                       // KDIM == 128 == 32 lanes * 4
  const float4 vd = *(const float4*)(last + (size_t)d * KDIM + f);
  const float4 vs = *(const float4*)(last + (size_t)s * KDIM + f);
  float* os = sum_n + (size_t)s * KDIM + f;
  float* od = sum_n + (size_t)d * KDIM + f;
  atomicAdd(os + 0, vd.x); atomicAdd(os + 1, vd.y);
  atomicAdd(os + 2, vd.z); atomicAdd(os + 3, vd.w);
  atomicAdd(od + 0, vs.x); atomicAdd(od + 1, vs.y);
  atomicAdd(od + 2, vs.z); atomicAdd(od + 3, vs.w);
}

// out[b,c] = concat(last[m], pooled[m]) @ clf_W + clf_b   (tiny)
__global__ void clf_kernel(const float* __restrict__ last,
                           const float* __restrict__ sum_n,
                           const float* __restrict__ cnt,
                           const int* __restrict__ mapping,
                           const float* __restrict__ Wc,   // [256, 40]
                           const float* __restrict__ bc,   // [40]
                           float* __restrict__ out) {
  const int idx = blockIdx.x * blockDim.x + threadIdx.x;
  if (idx >= BATCH * CDIM) return;
  const int b = idx / CDIM, c = idx % CDIM;
  const int m = mapping[b];
  const float inv = 1.0f / (1.0f + cnt[m]);
  float acc = bc[c];
  for (int j = 0; j < KDIM; ++j) {
    const float lv = last[(size_t)m * KDIM + j];
    const float pv = (lv + sum_n[(size_t)m * KDIM + j]) * inv;
    acc += lv * Wc[j * CDIM + c] + pv * Wc[(KDIM + j) * CDIM + c];
  }
  out[idx] = acc;
}

// ---------------------------------------------------------------------------
// Host-side orchestration
// ---------------------------------------------------------------------------
static inline unsigned int gblocks(long long total, int bs) {
  return (unsigned int)((total + bs - 1) / bs);
}

extern "C" void kernel_launch(void* const* d_in, const int* in_sizes, int n_in,
                              void* d_out, int out_size, void* d_ws, size_t ws_size,
                              hipStream_t stream) {
  const float* hidden  = (const float*)d_in[0];   // [L,N,1024]
  const float* pW1     = (const float*)d_in[1];   // [L,1024,128]
  const float* pb1     = (const float*)d_in[2];   // [L,128]
  const float* lng     = (const float*)d_in[3];   // [L,128]
  const float* lnb     = (const float*)d_in[4];   // [L,128]
  const float* pW2     = (const float*)d_in[5];   // [L,128,128]
  const float* pb2     = (const float*)d_in[6];   // [L,128]
  const float* alphas  = (const float*)d_in[7];   // [L]
  const float* gW1     = (const float*)d_in[8];   // [L,128,256]
  const float* gb1     = (const float*)d_in[9];   // [L,256]
  const float* gW2     = (const float*)d_in[10];  // [L,256,128]
  const float* gb2     = (const float*)d_in[11];  // [L,128]
  const float* cW      = (const float*)d_in[12];  // [256,40]
  const float* cb      = (const float*)d_in[13];  // [40]
  const int*   eidx    = (const int*)d_in[14];    // [2,E]
  const int*   mapping = (const int*)d_in[15];    // [B]
  const int* src = eidx;
  const int* dst = eidx + N_EDGES;
  (void)ws_size; (void)in_sizes; (void)n_in; (void)out_size;

  // Workspace carve (aliasing: cnt reuses deg; sum_n reuses h)
  char* ws = (char*)d_ws;
  size_t off = 0;
  auto carve = [&](size_t bytes) -> void* {
    void* p = ws + off;
    off += (bytes + 255) & ~(size_t)255;
    return p;
  };
  float* deg   = (float*)carve((size_t)N_NODES * 4);         // reused as cnt
  float* selfw = (float*)carve((size_t)N_NODES * 4);
  float* norm  = (float*)carve((size_t)N_EDGES * 4);
  float* h     = (float*)carve((size_t)N_NODES * KDIM * 4);  // reused as sum_n
  float* z     = (float*)carve((size_t)N_NODES * HDIM * 4);
  float* hw    = (float*)carve((size_t)N_NODES * HDIM * 4);
  float* last  = (float*)carve((size_t)N_NODES * KDIM * 4);
  unsigned int* bpW1[LLAYERS];
  unsigned int* bpW2[LLAYERS];
  unsigned int* bgW1[LLAYERS];
  unsigned int* bgW2[LLAYERS];
  for (int l = 0; l < LLAYERS; ++l) {
    bpW1[l] = (unsigned int*)carve((size_t)DLM  * KDIM * 2);
    bpW2[l] = (unsigned int*)carve((size_t)KDIM * KDIM * 2);
    bgW1[l] = (unsigned int*)carve((size_t)KDIM * HDIM * 2);
    bgW2[l] = (unsigned int*)carve((size_t)HDIM * KDIM * 2);
  }

  const int BS = 256;
  const unsigned int rowBlocks  = N_NODES / 16;   // 6250, exact
  const unsigned int edgeBlocks = gblocks(N_EDGES, 8);  // 1 wave per edge

  // ---- Pre-pack all weights into WMMA-B fragment order (f16) ----
  for (int l = 0; l < LLAYERS; ++l) {
    pack_b_kernel<<<gblocks((long long)DLM * KDIM / 2, BS), BS, 0, stream>>>(
        pW1 + (size_t)l * DLM * KDIM, bpW1[l], DLM, KDIM);
    pack_b_kernel<<<gblocks((long long)KDIM * KDIM / 2, BS), BS, 0, stream>>>(
        pW2 + (size_t)l * KDIM * KDIM, bpW2[l], KDIM, KDIM);
    pack_b_kernel<<<gblocks((long long)KDIM * HDIM / 2, BS), BS, 0, stream>>>(
        gW1 + (size_t)l * KDIM * HDIM, bgW1[l], KDIM, HDIM);
    pack_b_kernel<<<gblocks((long long)HDIM * KDIM / 2, BS), BS, 0, stream>>>(
        gW2 + (size_t)l * HDIM * KDIM, bgW2[l], HDIM, KDIM);
  }

  // ---- Graph normalization: deg, norm, self_w ----
  fill_kernel<<<gblocks(N_NODES, BS), BS, 0, stream>>>(deg, 1.0f, N_NODES);
  deg_kernel<<<gblocks(N_EDGES, BS), BS, 0, stream>>>(dst, deg, N_EDGES);
  norm_kernel<<<gblocks(N_EDGES, BS), BS, 0, stream>>>(src, dst, deg, norm, N_EDGES);
  selfw_kernel<<<gblocks(N_NODES, BS), BS, 0, stream>>>(deg, selfw, N_NODES);

  for (int layer = 0; layer < LLAYERS; ++layer) {
    const float* X = hidden + (size_t)layer * N_NODES * DLM;

    // Projector (fused Linear/LN/ReLU/Linear/blend) -> h [N,128]
    proj_fused_kernel<<<rowBlocks, BS, 0, stream>>>(
        X, bpW1[layer], pb1 + layer * KDIM,
        lng + layer * KDIM, lnb + layer * KDIM,
        bpW2[layer], pb2 + layer * KDIM,
        alphas, layer, last, h);

    // GCN conv1: hw = h @ W1 [128->256]; z = hw*self_w + b1 + scatter
    gemm_wmma_kernel<KDIM, HDIM, false><<<rowBlocks, BS, 0, stream>>>(h, bgW1[layer], hw);
    agg_init_kernel<HDIM><<<gblocks((long long)N_NODES * HDIM, BS), BS, 0, stream>>>(
        hw, selfw, gb1 + layer * HDIM, z, (long long)N_NODES * HDIM);
    edge_scatter_kernel<HDIM><<<edgeBlocks, BS, 0, stream>>>(
        src, dst, norm, hw, z, N_EDGES);

    // GCN conv2: hw = relu(z) @ W2 [256->128]; last = hw*self_w + b2 + scatter
    gemm_wmma_kernel<HDIM, KDIM, true><<<rowBlocks, BS, 0, stream>>>(z, bgW2[layer], hw);
    agg_init_kernel<KDIM><<<gblocks((long long)N_NODES * KDIM, BS), BS, 0, stream>>>(
        hw, selfw, gb2 + layer * KDIM, last, (long long)N_NODES * KDIM);
    edge_scatter_kernel<KDIM><<<edgeBlocks, BS, 0, stream>>>(
        src, dst, norm, hw, last, N_EDGES);
  }

  // ---- Pooling over {node} U {out-nbrs} U {in-nbrs} ----
  float* cnt   = deg;   // deg dead, reuse
  float* sum_n = h;     // h dead, reuse
  fill_kernel<<<gblocks(N_NODES, BS), BS, 0, stream>>>(cnt, 0.0f, N_NODES);
  fill_kernel<<<gblocks((long long)N_NODES * KDIM, BS), BS, 0, stream>>>(
      sum_n, 0.0f, (long long)N_NODES * KDIM);
  cnt_kernel<<<gblocks(N_EDGES, BS), BS, 0, stream>>>(src, dst, cnt, N_EDGES);
  pool_scatter_kernel<<<edgeBlocks, BS, 0, stream>>>(src, dst, last, sum_n, N_EDGES);

  // ---- Classifier ----
  clf_kernel<<<gblocks(BATCH * CDIM, 128), 128, 0, stream>>>(
      last, sum_n, cnt, mapping, cW, cb, (float*)d_out);
}
